// MoEClassifier_88510686036634
// MI455X (gfx1250) — matrix operand order repaired and checked by reading, not compiled, not run on hardware
//
#include <hip/hip_runtime.h>
#include <hip/hip_bf16.h>

typedef __attribute__((ext_vector_type(16))) _Float16 v16h;
typedef __attribute__((ext_vector_type(8)))  _Float16 v8h;
typedef __attribute__((ext_vector_type(4)))  _Float16 v4h;
typedef __attribute__((ext_vector_type(8)))  float    v8f;
typedef _Float16 f16;

#define B_SZ  4096
#define D_SZ  1024
#define E_SZ  8
#define H_SZ  1024
#define O_SZ  512
#define HS_SZ 2048
#define M_SZ  64
#define CAP   4096   // per-expert row capacity (worst case = B)

// ---------------------------------------------------------------------------
// CDNA5 async global->LDS staging (ASYNCcnt path, ISA §15.18.3 / §5 ch.4)
// ---------------------------------------------------------------------------
__device__ __forceinline__ void async_copy16(const f16* gsrc, f16* ldst) {
  unsigned lds = (unsigned)(size_t)ldst;  // flat LDS aperture: low 32 = offset
  unsigned long long ga = (unsigned long long)(size_t)gsrc;
  asm volatile("global_load_async_to_lds_b128 %0, %1, off"
               :
               : "v"(lds), "v"(ga)
               : "memory");
}

__device__ __forceinline__ void async_wait0() {
  asm volatile("s_wait_asynccnt 0" ::: "memory");
}

// ---------------------------------------------------------------------------
// WMMA fragment helpers (CDNA5 ISA §7.12.2). All loads are 16B contiguous.
// ---------------------------------------------------------------------------
__device__ __forceinline__ v16h concat8(v8h lo, v8h hi) {
  return __builtin_shufflevector(lo, hi, 0, 1, 2, 3, 4, 5, 6, 7, 8, 9, 10, 11,
                                 12, 13, 14, 15);
}

// A row-major [m][k]: lane m = lid&15; elems 0..7 = k 8h..8h+7,
// elems 8..15 = k 16+8h..+7  -> two contiguous 16B LDS loads.
__device__ __forceinline__ v16h load_a_frag(const f16* Abase, int lda,
                                            int lid) {
  int l = lid & 15, h = lid >> 4;
  const f16* r = Abase + l * lda + 8 * h;
  v8h lo = *(const v8h*)(r);
  v8h hi = *(const v8h*)(r + 16);
  return concat8(lo, hi);
}

// B stored TRANSPOSED: Bt[n][k]. Lane n = lid&15; elems 0..7 = k 8h..8h+7, 8..15 = k 16+8h..+7
__device__ __forceinline__ v16h load_bt_frag(const f16* Btbase, int ldbt,
                                             int lid) {
  int l = lid & 15, h = lid >> 4;
  const f16* r = Btbase + l * ldbt + 8 * h;
  v8h lo = *(const v8h*)(r);
  v8h hi = *(const v8h*)(r + 16);
  return concat8(lo, hi);
}

__device__ __forceinline__ v8f wmma16(v16h a, v16h b, v8f c) {
  return __builtin_amdgcn_wmma_f32_16x16x32_f16(false, a, false, b, (short)0, c,
                                                false, false);
}

// Stage 256(n) x 32(k) tile of pre-transposed f16 weights into Bt[256][40].
// Pure contiguous copies -> async global->LDS, 4 x 16B per thread.
__device__ __forceinline__ void stage_bt_async(const f16* __restrict__ Wt,
                                               int ldk, int n0, int k0,
                                               f16* __restrict__ Bt, int tid) {
#pragma unroll
  for (int i = 0; i < 4; ++i) {
    int q = i * 256 + tid;  // 0..1023 chunk id, 16B each
    int n = q >> 2;
    int koff = (q & 3) * 8;
    async_copy16(Wt + (size_t)(n0 + n) * ldk + k0 + koff, Bt + n * 40 + koff);
  }
}

// ---------------------------------------------------------------------------
// Kernel: fp32 [K][N] -> f16 transposed [N][K] (LDS-tiled 32x32)
// grid: (N/32, K/32, nmat), matrix stride K*N
// ---------------------------------------------------------------------------
__global__ __launch_bounds__(256) void cvt_t_kernel(
    const float* __restrict__ src, f16* __restrict__ dst, int K, int N) {
  __shared__ f16 T[32][36];
  size_t mat = (size_t)blockIdx.z * K * N;
  const float* S = src + mat;
  f16* Dt = dst + mat;
  int k0 = blockIdx.y * 32, n0 = blockIdx.x * 32;
  int t = threadIdx.x;
  {
    int r = t >> 3, c = (t & 7) * 4;
    float4 f = *(const float4*)(S + (size_t)(k0 + r) * N + n0 + c);
    T[r][c + 0] = (f16)f.x;
    T[r][c + 1] = (f16)f.y;
    T[r][c + 2] = (f16)f.z;
    T[r][c + 3] = (f16)f.w;
  }
  __syncthreads();
  {
    int n = t >> 3, kq = (t & 7) * 4;
    v4h p = {T[kq][n], T[kq + 1][n], T[kq + 2][n], T[kq + 3][n]};
    *(v4h*)(Dt + (size_t)(n0 + n) * K + k0 + kq) = p;
  }
}

// ---------------------------------------------------------------------------
// Kernel: zero bucket counters
// ---------------------------------------------------------------------------
__global__ void init_counts(int* counts) {
  if (threadIdx.x < E_SZ) counts[threadIdx.x] = 0;
}

// ---------------------------------------------------------------------------
// Kernel: gather x (fp32 -> f16), gating, top-2, softmax, bucket scatter
// ---------------------------------------------------------------------------
__global__ __launch_bounds__(256) void gate_kernel(
    const int* __restrict__ prompt_ids, const float* __restrict__ prompt_table,
    const float* __restrict__ Wg, const float* __restrict__ bg,
    f16* __restrict__ xh, int* __restrict__ counts, int* __restrict__ rowlist,
    float* __restrict__ wlist) {
  __shared__ float sW[8][8];
  __shared__ float lg[8];
  int row = blockIdx.x;
  int tid = threadIdx.x, lid = tid & 31, w = tid >> 5;
  int pid = prompt_ids[row];
  const float* xr = prompt_table + (size_t)pid * D_SZ;

  float accE[8] = {0, 0, 0, 0, 0, 0, 0, 0};
  for (int i = tid; i < D_SZ; i += 256) {
    float xv = xr[i];
    xh[(size_t)row * D_SZ + i] = (f16)xv;
#pragma unroll
    for (int e = 0; e < 8; ++e) accE[e] += xv * Wg[i * 8 + e];
  }
#pragma unroll
  for (int e = 0; e < 8; ++e) {
    float v = accE[e];
#pragma unroll
    for (int o = 16; o > 0; o >>= 1) v += __shfl_xor(v, o);
    if (lid == 0) sW[w][e] = v;
  }
  __syncthreads();
  if (tid < 8) {
    float s = bg[tid];
#pragma unroll
    for (int ww = 0; ww < 8; ++ww) s += sW[ww][tid];
    lg[tid] = s;
  }
  __syncthreads();
  if (tid == 0) {
    int i0 = 0; float v0 = lg[0];
    for (int e = 1; e < 8; ++e) if (lg[e] > v0) { v0 = lg[e]; i0 = e; }
    int i1 = -1; float v1 = -3.4e38f;
    for (int e = 0; e < 8; ++e)
      if (e != i0 && lg[e] > v1) { v1 = lg[e]; i1 = e; }
    float e1 = expf(v1 - v0);
    float w0 = 1.f / (1.f + e1);
    float w1 = e1 / (1.f + e1);
    int p0 = atomicAdd(&counts[i0], 1);
    rowlist[i0 * CAP + p0] = row; wlist[i0 * CAP + p0] = w0;
    int p1 = atomicAdd(&counts[i1], 1);
    rowlist[i1 * CAP + p1] = row; wlist[i1 * CAP + p1] = w1;
  }
}

// ---------------------------------------------------------------------------
// Kernel: s1h = relu(x @ Ws1 + bs1) f16.  M=4096 N=2048 K=1024
// Double-buffered async LDS pipeline; wave tile 32x64 (2 A-frags x 4 B-frags)
// ---------------------------------------------------------------------------
__global__ __launch_bounds__(256) void gemm_s1(
    const f16* __restrict__ xh, const f16* __restrict__ Wt,
    const float* __restrict__ bs1, f16* __restrict__ s1h) {
  __shared__ f16 As[2][64][40];
  __shared__ f16 Bt[2][256][40];
  int tid = threadIdx.x, lid = tid & 31, w = tid >> 5;
  int m0 = blockIdx.y * 64, n0 = blockIdx.x * 256;
  int mg = (w & 1) * 32;
  int ng = (w >> 1) * 64;
  int ar = tid >> 2, ac = (tid & 3) * 8;

  v8f acc[2][4];
#pragma unroll
  for (int mi = 0; mi < 2; ++mi)
#pragma unroll
    for (int s = 0; s < 4; ++s) acc[mi][s] = (v8f){0, 0, 0, 0, 0, 0, 0, 0};

  // preload tile 0
  async_copy16(xh + (size_t)(m0 + ar) * D_SZ + ac, &As[0][ar][ac]);
  stage_bt_async(Wt, D_SZ, n0, 0, &Bt[0][0][0], tid);

  int t = 0;
  for (int k0 = 0; k0 < D_SZ; k0 += 32, t ^= 1) {
    async_wait0();
    __syncthreads();
    if (k0 + 32 < D_SZ) {  // prefetch next tile into other buffer
      async_copy16(xh + (size_t)(m0 + ar) * D_SZ + k0 + 32 + ac,
                   &As[t ^ 1][ar][ac]);
      stage_bt_async(Wt, D_SZ, n0, k0 + 32, &Bt[t ^ 1][0][0], tid);
    }
    v16h a0 = load_a_frag(&As[t][mg][0], 40, lid);
    v16h a1 = load_a_frag(&As[t][mg + 16][0], 40, lid);
#pragma unroll
    for (int s = 0; s < 4; ++s) {
      v16h b = load_bt_frag(&Bt[t][ng + s * 16][0], 40, lid);
      acc[0][s] = wmma16(a0, b, acc[0][s]);
      acc[1][s] = wmma16(a1, b, acc[1][s]);
    }
  }
  int l = lid & 15, hh = lid >> 4;
#pragma unroll
  for (int mi = 0; mi < 2; ++mi)
#pragma unroll
    for (int s = 0; s < 4; ++s) {
      int ncol = n0 + ng + s * 16 + l;
      float bias = bs1[ncol];
#pragma unroll
      for (int v = 0; v < 8; ++v) {
        int mrow = m0 + mg + mi * 16 + v + 8 * hh;
        float val = acc[mi][s][v] + bias;
        val = val > 0.f ? val : 0.f;
        s1h[(size_t)mrow * HS_SZ + ncol] = (f16)val;
      }
    }
}

// ---------------------------------------------------------------------------
// Kernel: hq = s1h @ Ws2 + bs2 (fp32 out). M=4096 N=512 K=2048, Wt ldk=HS
// ---------------------------------------------------------------------------
__global__ __launch_bounds__(256) void gemm_s2(
    const f16* __restrict__ s1h, const f16* __restrict__ Wt,
    const float* __restrict__ bs2, float* __restrict__ hq) {
  __shared__ f16 As[2][64][40];
  __shared__ f16 Bt[2][256][40];
  int tid = threadIdx.x, lid = tid & 31, w = tid >> 5;
  int m0 = blockIdx.y * 64, n0 = blockIdx.x * 256;
  int mg = (w & 1) * 32;
  int ng = (w >> 1) * 64;
  int ar = tid >> 2, ac = (tid & 3) * 8;

  v8f acc[2][4];
#pragma unroll
  for (int mi = 0; mi < 2; ++mi)
#pragma unroll
    for (int s = 0; s < 4; ++s) acc[mi][s] = (v8f){0, 0, 0, 0, 0, 0, 0, 0};

  async_copy16(s1h + (size_t)(m0 + ar) * HS_SZ + ac, &As[0][ar][ac]);
  stage_bt_async(Wt, HS_SZ, n0, 0, &Bt[0][0][0], tid);

  int t = 0;
  for (int k0 = 0; k0 < HS_SZ; k0 += 32, t ^= 1) {
    async_wait0();
    __syncthreads();
    if (k0 + 32 < HS_SZ) {
      async_copy16(s1h + (size_t)(m0 + ar) * HS_SZ + k0 + 32 + ac,
                   &As[t ^ 1][ar][ac]);
      stage_bt_async(Wt, HS_SZ, n0, k0 + 32, &Bt[t ^ 1][0][0], tid);
    }
    v16h a0 = load_a_frag(&As[t][mg][0], 40, lid);
    v16h a1 = load_a_frag(&As[t][mg + 16][0], 40, lid);
#pragma unroll
    for (int s = 0; s < 4; ++s) {
      v16h b = load_bt_frag(&Bt[t][ng + s * 16][0], 40, lid);
      acc[0][s] = wmma16(a0, b, acc[0][s]);
      acc[1][s] = wmma16(a1, b, acc[1][s]);
    }
  }
  int l = lid & 15, hh = lid >> 4;
#pragma unroll
  for (int mi = 0; mi < 2; ++mi)
#pragma unroll
    for (int s = 0; s < 4; ++s) {
      int ncol = n0 + ng + s * 16 + l;
      float bias = bs2[ncol];
#pragma unroll
      for (int v = 0; v < 8; ++v) {
        int mrow = m0 + mg + mi * 16 + v + 8 * hh;
        hq[(size_t)mrow * O_SZ + ncol] = acc[mi][s][v] + bias;
      }
    }
}

// ---------------------------------------------------------------------------
// Kernel: fused grouped expert FFN. grid = (CAP/64, E).
// Phase A: Hs(64x1024 f16, LDS) = relu(X_e @ W1[e] + b1[e])
// Phase B: Y(64x512) = Hs @ W2[e] + b2[e]; hq += gate_w * Y (atomics)
// Double-buffered async staging. LDS: Hs | As[2] | Bt[2] | rows | wgt
// ---------------------------------------------------------------------------
__global__ __launch_bounds__(256) void expert_kernel(
    const f16* __restrict__ xh, const f16* __restrict__ W1t,
    const float* __restrict__ b1, const f16* __restrict__ W2t,
    const float* __restrict__ b2, const int* __restrict__ counts,
    const int* __restrict__ rowlist, const float* __restrict__ wlist,
    float* __restrict__ hq) {
  extern __shared__ char smem[];
  f16* Hs = (f16*)smem;                   // 64 x 1032
  f16* As = Hs + 64 * 1032;               // 2 x 64 x 40
  f16* Bt = As + 2 * 64 * 40;             // 2 x 256 x 40
  int* rows = (int*)(Bt + 2 * 256 * 40);  // 64
  float* wgt = (float*)(rows + 64);       // 64

  int e = blockIdx.y;
  int cnt = counts[e];
  int s0 = blockIdx.x * 64;
  if (s0 >= cnt) return;  // uniform across block

  int tid = threadIdx.x, lid = tid & 31, w = tid >> 5;
  int mg = (w & 1) * 32;
  int ng = (w >> 1) * 64;
  int l = lid & 15, hh = lid >> 4;
  int ar = tid >> 2, ac = (tid & 3) * 8;

  if (tid < 64) {
    int slot = s0 + tid;
    rows[tid] = (slot < cnt) ? rowlist[e * CAP + slot] : rowlist[e * CAP];
    wgt[tid] = (slot < cnt) ? wlist[e * CAP + slot] : 0.f;
  }
  __syncthreads();
  int myrow = rows[ar];  // gather row for this thread's A-staging chunk

  const f16* W1e = W1t + (size_t)e * D_SZ * H_SZ;
  const f16* W2e = W2t + (size_t)e * H_SZ * O_SZ;

  // ---- Phase A: h tile into LDS ----
  for (int nblk = 0; nblk < H_SZ; nblk += 256) {
    v8f acc[2][4];
#pragma unroll
    for (int mi = 0; mi < 2; ++mi)
#pragma unroll
      for (int s = 0; s < 4; ++s) acc[mi][s] = (v8f){0, 0, 0, 0, 0, 0, 0, 0};

    __syncthreads();  // all waves done reading buffers from previous nblk
    async_copy16(xh + (size_t)myrow * D_SZ + ac, As + ar * 40 + ac);
    stage_bt_async(W1e, D_SZ, nblk, 0, Bt, tid);

    int t = 0;
    for (int k0 = 0; k0 < D_SZ; k0 += 32, t ^= 1) {
      async_wait0();
      __syncthreads();
      if (k0 + 32 < D_SZ) {
        async_copy16(xh + (size_t)myrow * D_SZ + k0 + 32 + ac,
                     As + (t ^ 1) * 64 * 40 + ar * 40 + ac);
        stage_bt_async(W1e, D_SZ, nblk, k0 + 32, Bt + (t ^ 1) * 256 * 40, tid);
      }
      v16h a0 = load_a_frag(As + t * 64 * 40 + mg * 40, 40, lid);
      v16h a1 = load_a_frag(As + t * 64 * 40 + (mg + 16) * 40, 40, lid);
#pragma unroll
      for (int s = 0; s < 4; ++s) {
        v16h b =
            load_bt_frag(Bt + t * 256 * 40 + (ng + s * 16) * 40, 40, lid);
        acc[0][s] = wmma16(a0, b, acc[0][s]);
        acc[1][s] = wmma16(a1, b, acc[1][s]);
      }
    }
#pragma unroll
    for (int mi = 0; mi < 2; ++mi)
#pragma unroll
      for (int s = 0; s < 4; ++s) {
        int ncol = nblk + ng + s * 16 + l;
        float bias = b1[e * H_SZ + ncol];
#pragma unroll
        for (int v = 0; v < 8; ++v) {
          int mloc = mg + mi * 16 + v + 8 * hh;
          float val = acc[mi][s][v] + bias;
          val = val > 0.f ? val : 0.f;
          Hs[mloc * 1032 + ncol] = (f16)val;
        }
      }
  }
  __syncthreads();

  // ---- Phase B: Y = Hs @ W2[e]; weighted atomic combine into hq ----
  for (int nblk = 0; nblk < O_SZ; nblk += 256) {
    v8f acc[2][4];
#pragma unroll
    for (int mi = 0; mi < 2; ++mi)
#pragma unroll
      for (int s = 0; s < 4; ++s) acc[mi][s] = (v8f){0, 0, 0, 0, 0, 0, 0, 0};

    __syncthreads();
    stage_bt_async(W2e, H_SZ, nblk, 0, Bt, tid);

    int t = 0;
    for (int k0 = 0; k0 < H_SZ; k0 += 32, t ^= 1) {
      async_wait0();
      __syncthreads();
      if (k0 + 32 < H_SZ)
        stage_bt_async(W2e, H_SZ, nblk, k0 + 32, Bt + (t ^ 1) * 256 * 40, tid);
      v16h a0 = load_a_frag(Hs + mg * 1032 + k0, 1032, lid);
      v16h a1 = load_a_frag(Hs + (mg + 16) * 1032 + k0, 1032, lid);
#pragma unroll
      for (int s = 0; s < 4; ++s) {
        v16h b =
            load_bt_frag(Bt + t * 256 * 40 + (ng + s * 16) * 40, 40, lid);
        acc[0][s] = wmma16(a0, b, acc[0][s]);
        acc[1][s] = wmma16(a1, b, acc[1][s]);
      }
    }
#pragma unroll
    for (int mi = 0; mi < 2; ++mi)
#pragma unroll
      for (int s = 0; s < 4; ++s) {
        int ncol = nblk + ng + s * 16 + l;
        float bias = b2[e * O_SZ + ncol];
#pragma unroll
        for (int v = 0; v < 8; ++v) {
          int mloc = mg + mi * 16 + v + 8 * hh;
          if (s0 + mloc < cnt) {
            float val = (acc[mi][s][v] + bias) * wgt[mloc];
            atomicAdd(&hq[(size_t)rows[mloc] * O_SZ + ncol], val);
          }
        }
      }
  }
}

// ---------------------------------------------------------------------------
// Kernel: head.  logit = (hq@Wa + ba)·theta - (hq·Wd + bd). wave per row.
// ---------------------------------------------------------------------------
__global__ __launch_bounds__(256) void head_kernel(
    const float* __restrict__ hq, const float* __restrict__ Wd,
    const float* __restrict__ bd, const float* __restrict__ Wa,
    const float* __restrict__ ba, const float* __restrict__ model_table,
    const int* __restrict__ model_ids, float* __restrict__ out) {
  int tid = threadIdx.x, lid = tid & 31, w = tid >> 5;
  int row = blockIdx.x * 8 + w;
  const float* h = hq + (size_t)row * O_SZ;

  float hv[16];
#pragma unroll
  for (int i = 0; i < 16; ++i) hv[i] = h[lid * 16 + i];

  float am[64];
#pragma unroll
  for (int m = 0; m < 64; ++m) am[m] = 0.f;
#pragma unroll
  for (int i = 0; i < 16; ++i) {
    int o = lid * 16 + i;
    float x = hv[i];
    const float* war = Wa + (size_t)o * M_SZ;
#pragma unroll
    for (int m = 0; m < 64; ++m) am[m] += x * war[m];
  }
  const float* th = model_table + (size_t)model_ids[row] * M_SZ;
  float c = 0.f, bq = 0.f;
#pragma unroll
  for (int i = 0; i < 16; ++i) bq += hv[i] * Wd[lid * 16 + i];
#pragma unroll
  for (int m = 0; m < 64; ++m) c += am[m] * th[m];
#pragma unroll
  for (int o = 16; o > 0; o >>= 1) {
    c += __shfl_xor(c, o);
    bq += __shfl_xor(bq, o);
  }
  if (lid == 0) {
    float bt = 0.f;
    for (int m = 0; m < 64; ++m) bt += ba[m] * th[m];
    out[row] = c + bt - (bq + bd[0]);
  }
}

// ---------------------------------------------------------------------------
extern "C" void kernel_launch(void* const* d_in, const int* in_sizes, int n_in,
                              void* d_out, int out_size, void* d_ws,
                              size_t ws_size, hipStream_t stream) {
  const int*   model_ids  = (const int*)d_in[0];
  const int*   prompt_ids = (const int*)d_in[1];
  const float* prompt_tab = (const float*)d_in[2];
  const float* model_tab  = (const float*)d_in[3];
  const float* Wg  = (const float*)d_in[4];
  const float* bg  = (const float*)d_in[5];
  const float* W1  = (const float*)d_in[6];
  const float* b1  = (const float*)d_in[7];
  const float* W2  = (const float*)d_in[8];
  const float* b2  = (const float*)d_in[9];
  const float* Ws1 = (const float*)d_in[10];
  const float* bs1 = (const float*)d_in[11];
  const float* Ws2 = (const float*)d_in[12];
  const float* bs2 = (const float*)d_in[13];
  const float* Wd  = (const float*)d_in[14];
  const float* bd  = (const float*)d_in[15];
  const float* Wa  = (const float*)d_in[16];
  const float* ba  = (const float*)d_in[17];
  float* out = (float*)d_out;
  (void)in_sizes; (void)n_in; (void)out_size; (void)ws_size;

  char* ws = (char*)d_ws;
  size_t off = 0;
  auto alloc = [&](size_t bytes) -> void* {
    void* p = ws + off;
    off = (off + bytes + 255) & ~(size_t)255;
    return p;
  };
  f16*   xh      = (f16*)alloc((size_t)B_SZ * D_SZ * 2);
  f16*   s1h     = (f16*)alloc((size_t)B_SZ * HS_SZ * 2);
  float* hq      = (float*)alloc((size_t)B_SZ * O_SZ * 4);
  int*   counts  = (int*)alloc(E_SZ * 4);
  int*   rowlist = (int*)alloc((size_t)E_SZ * CAP * 4);
  float* wlist   = (float*)alloc((size_t)E_SZ * CAP * 4);
  f16*   w1t     = (f16*)alloc((size_t)E_SZ * D_SZ * H_SZ * 2);  // [E][H][D]
  f16*   w2t     = (f16*)alloc((size_t)E_SZ * H_SZ * O_SZ * 2);  // [E][O][H]
  f16*   ws1t    = (f16*)alloc((size_t)D_SZ * HS_SZ * 2);        // [HS][D]
  f16*   ws2t    = (f16*)alloc((size_t)HS_SZ * O_SZ * 2);        // [O][HS]

  // one-shot fp32 -> f16 transpose of all weight matrices
  cvt_t_kernel<<<dim3(H_SZ / 32, D_SZ / 32, E_SZ), 256, 0, stream>>>(W1, w1t,
                                                                     D_SZ,
                                                                     H_SZ);
  cvt_t_kernel<<<dim3(O_SZ / 32, H_SZ / 32, E_SZ), 256, 0, stream>>>(W2, w2t,
                                                                     H_SZ,
                                                                     O_SZ);
  cvt_t_kernel<<<dim3(HS_SZ / 32, D_SZ / 32, 1), 256, 0, stream>>>(Ws1, ws1t,
                                                                   D_SZ,
                                                                   HS_SZ);
  cvt_t_kernel<<<dim3(O_SZ / 32, HS_SZ / 32, 1), 256, 0, stream>>>(Ws2, ws2t,
                                                                   HS_SZ,
                                                                   O_SZ);

  init_counts<<<1, 32, 0, stream>>>(counts);
  gate_kernel<<<B_SZ, 256, 0, stream>>>(prompt_ids, prompt_tab, Wg, bg, xh,
                                        counts, rowlist, wlist);
  gemm_s1<<<dim3(HS_SZ / 256, B_SZ / 64), 256, 0, stream>>>(xh, ws1t, bs1,
                                                            s1h);
  gemm_s2<<<dim3(O_SZ / 256, B_SZ / 64), 256, 0, stream>>>(s1h, ws2t, bs2, hq);
  size_t smem =
      (size_t)(64 * 1032 + 2 * 64 * 40 + 2 * 256 * 40) * 2 + 64 * 4 + 64 * 4;
  expert_kernel<<<dim3(CAP / 64, E_SZ), 256, smem, stream>>>(
      xh, w1t, b1, w2t, b2, counts, rowlist, wlist, hq);
  head_kernel<<<B_SZ / 8, 256, 0, stream>>>(hq, Wd, bd, Wa, ba, model_tab,
                                            model_ids, out);
}
